// EfficientVectorRouting_30623116821356
// MI455X (gfx1250) — matrix operand order repaired and checked by reading, not compiled, not run on hardware
//
#include <hip/hip_runtime.h>
#include <stdint.h>

// ---------------------------------------------------------------------------
// EfficientVectorRouting for MI455X (gfx1250, wave32, WMMA).
// All K>=1024 reductions go through v_wmma_f32_16x16x32_bf16 (f32 accum).
// Fragment layouts follow CDNA5 ISA 7.12.2 (A 16x32, B 32x16, C/D 16x16 f32).
// Every bf16 A/B fragment is two b128 loads per lane; an f32 B fragment is four.
// The dominant GEMM (t = phiT x XsT, K=4096) is K-split 4 ways into private
// partial buffers (deterministic; folded in the tw elementwise kernel).
// ---------------------------------------------------------------------------

typedef __attribute__((ext_vector_type(16))) __bf16   v16bf;
typedef __attribute__((ext_vector_type(8)))  float    v8f;
typedef __attribute__((ext_vector_type(4)))  uint32_t u32x4;
typedef __attribute__((ext_vector_type(4)))  float    f32x4;

#define B_       8
#define N_INP_   4096
#define N_OUT_   64
#define D_       1024
#define KCHUNK_  1024          // every GEMM K-slice is exactly 1024
#define TPART_   ((long)B_ * N_OUT_ * D_)   // one t-partial buffer

__device__ inline unsigned short f2bf(float f) {
  union { float f; uint32_t u; } c; c.f = f;
  uint32_t u = c.u;
  uint32_t r = u + 0x7FFFu + ((u >> 16) & 1u);   // round to nearest even
  return (unsigned short)(r >> 16);
}
__device__ inline float bf2f(unsigned short s) {
  union { uint32_t u; float f; } c; c.u = ((uint32_t)s) << 16;
  return c.f;
}

union Frag { uint32_t w[8]; u32x4 q[2]; v16bf v; };

__device__ inline void loadFragBf(Frag& f, const unsigned short* p) {
  f.q[0] = *(const u32x4*)p;
  f.q[1] = *(const u32x4*)(p + 16);
}
__device__ inline void loadFragF32(Frag& f, const float* p) {
  const f32x4* p4 = (const f32x4*)p;
  f32x4 x0 = p4[0], x1 = p4[1], x2 = p4[4], x3 = p4[5];
  float e[16];
#pragma unroll
  for (int c = 0; c < 4; ++c) {
    e[c] = x0[c]; e[4 + c] = x1[c]; e[8 + c] = x2[c]; e[12 + c] = x3[c];
  }
#pragma unroll
  for (int v = 0; v < 8; ++v)
    f.w[v] = (uint32_t)f2bf(e[2 * v]) | ((uint32_t)f2bf(e[2 * v + 1]) << 16);
}

// -------------------------- small kernels ----------------------------------

// out[r][c] = bf16(in[c][r]); 1024x1024
__global__ void transpose_bf16_kernel(const float* __restrict__ in,
                                      unsigned short* __restrict__ out) {
  int idx = blockIdx.x * 256 + threadIdx.x;
  int r = idx >> 10, c = idx & 1023;
  out[idx] = f2bf(in[c * 1024 + r]);
}

// Per row (b,i): write XsT[b][d][i] = bf16(x/64), f_a[b,i]=sigmoid(dot+B_A)
__global__ void rowstats_kernel(const float* __restrict__ x,
                                const float* __restrict__ W_A,
                                const float* __restrict__ B_A,
                                unsigned short* __restrict__ XsT,
                                float* __restrict__ f_a) {
  int row = blockIdx.x;                // b*4096 + i
  int b = row >> 12, i = row & 4095;
  const float* xr = x + (long)row * D_;
  const float* wr = W_A + (long)i * D_;
  float acc = 0.f;
#pragma unroll
  for (int q = 0; q < 4; ++q) {
    int d = threadIdx.x + q * 256;
    float xs = xr[d] * 0.015625f;      // N_INP^-0.5 = 1/64
    acc += xs * wr[d];
    XsT[((long)b * D_ + d) * N_INP_ + i] = f2bf(xs);
  }
  __shared__ float red[256];
  red[threadIdx.x] = acc; __syncthreads();
  for (int s = 128; s > 0; s >>= 1) {
    if (threadIdx.x < s) red[threadIdx.x] += red[threadIdx.x + s];
    __syncthreads();
  }
  if (threadIdx.x == 0)
    f_a[row] = 1.f / (1.f + __expf(-(red[0] + B_A[i])));
}

// iter-0 phi: R = 1/64 uniform. phiT[b][j][i] (bf16)
__global__ void phi0_kernel(const float* __restrict__ f_a,
                            const float* __restrict__ bu,
                            const float* __restrict__ bi,
                            unsigned short* __restrict__ phiT) {
  int bj = blockIdx.x;                 // b*64 + j
  int b = bj >> 6, j = bj & 63;
#pragma unroll
  for (int q = 0; q < 16; ++q) {
    int i = threadIdx.x + q * 256;
    float fa  = f_a[b * N_INP_ + i];
    float phi = fa * (bu[i * 64 + j] * (1.f / 64.f) - bi[i * 64 + j] * (63.f / 64.f));
    phiT[(long)bj * N_INP_ + i] = f2bf(phi);
  }
}

// phisum[b][j] = sum_i phiT[b][j][i]
__global__ void phisum_kernel(const unsigned short* __restrict__ phiT,
                              float* __restrict__ phisum) {
  int bj = blockIdx.x;
  float acc = 0.f;
#pragma unroll
  for (int q = 0; q < 16; ++q) {
    int i = threadIdx.x + q * 256;
    acc += bf2f(phiT[(long)bj * N_INP_ + i]);
  }
  __shared__ float red[256];
  red[threadIdx.x] = acc; __syncthreads();
  for (int s = 128; s > 0; s >>= 1) {
    if (threadIdx.x < s) red[threadIdx.x] += red[threadIdx.x + s];
    __syncthreads();
  }
  if (threadIdx.x == 0) phisum[bj] = red[0];
}

// tw[b][j][d] = bf16((sum of 4 K-partials of t) * W_F1[j][d])
__global__ void tw_kernel(const float* __restrict__ tpart,
                          const float* __restrict__ W_F1,
                          unsigned short* __restrict__ tw) {
  long idx = (long)blockIdx.x * 256 + threadIdx.x;   // 8*64*1024
  int jd = (int)(idx & (N_OUT_ * D_ - 1));
  float s = tpart[idx] + tpart[idx + TPART_] +
            tpart[idx + 2 * TPART_] + tpart[idx + 3 * TPART_];
  tw[idx] = f2bf(s * W_F1[jd]);
}

// LayerNorm over last dim (1024). One block per row (b*64 rows).
__global__ void ln_kernel(const float* __restrict__ x,
                          unsigned short* __restrict__ outBf,
                          float* __restrict__ outF) {
  int row = blockIdx.x;
  const float* xr = x + (long)row * D_;
  float v[4], s = 0.f, ss = 0.f;
#pragma unroll
  for (int q = 0; q < 4; ++q) {
    int d = threadIdx.x + q * 256;
    v[q] = xr[d]; s += v[q]; ss += v[q] * v[q];
  }
  __shared__ float red[256];
  red[threadIdx.x] = s; __syncthreads();
  for (int st = 128; st > 0; st >>= 1) {
    if (threadIdx.x < st) red[threadIdx.x] += red[threadIdx.x + st];
    __syncthreads();
  }
  float S = red[0]; __syncthreads();
  red[threadIdx.x] = ss; __syncthreads();
  for (int st = 128; st > 0; st >>= 1) {
    if (threadIdx.x < st) red[threadIdx.x] += red[threadIdx.x + st];
    __syncthreads();
  }
  float SS = red[0];
  float mu = S * (1.f / D_);
  float var = SS * (1.f / D_) - mu * mu;
  float rs = rsqrtf(var + 1e-5f);
#pragma unroll
  for (int q = 0; q < 4; ++q) {
    int d = threadIdx.x + q * 256;
    float y = (v[q] - mu) * rs;
    if (outBf) outBf[(long)row * D_ + d] = f2bf(y);
    if (outF)  outF[(long)row * D_ + d]  = y;
  }
}

// softmax over j (64) + phi for iter 1. One wave per (b,i) row.
__global__ void softmax_phi_kernel(const float* __restrict__ SbufT,
                                   const float* __restrict__ W_S,
                                   const float* __restrict__ B_S,
                                   const float* __restrict__ f_a,
                                   const float* __restrict__ bu,
                                   const float* __restrict__ bi,
                                   unsigned short* __restrict__ phiT) {
  int wave = threadIdx.x >> 5, lane = threadIdx.x & 31;
  int row = blockIdx.x * 8 + wave;     // b*4096 + i
  int b = row >> 12, i = row & 4095;
  int j0 = lane, j1 = lane + 32;
  float z0 = SbufT[((long)b * N_OUT_ + j0) * N_INP_ + i];
  float z1 = SbufT[((long)b * N_OUT_ + j1) * N_INP_ + i];
  float zz0 = z0 * W_S[i * 64 + j0] + B_S[i * 64 + j0];
  float zz1 = z1 * W_S[i * 64 + j1] + B_S[i * 64 + j1];
  float s0 = fminf(zz0, 0.f) - log1pf(__expf(-fabsf(zz0)));
  float s1 = fminf(zz1, 0.f) - log1pf(__expf(-fabsf(zz1)));
  float m = fmaxf(s0, s1);
  for (int o = 16; o > 0; o >>= 1) m = fmaxf(m, __shfl_xor(m, o, 32));
  float e0 = __expf(s0 - m), e1 = __expf(s1 - m);
  float e = e0 + e1;
  for (int o = 16; o > 0; o >>= 1) e += __shfl_xor(e, o, 32);
  float inv = 1.f / e;
  float fa = f_a[row];
  float R0 = e0 * inv, R1 = e1 * inv;
  float bu0 = bu[i * 64 + j0], bi0 = bi[i * 64 + j0];
  float bu1 = bu[i * 64 + j1], bi1 = bi[i * 64 + j1];
  float p0 = fa * ((bu0 + bi0) * R0 - bi0);
  float p1 = fa * ((bu1 + bi1) * R1 - bi1);
  phiT[((long)b * N_OUT_ + j0) * N_INP_ + i] = f2bf(p0);
  phiT[((long)b * N_OUT_ + j1) * N_INP_ + i] = f2bf(p1);
}

// -------------------------- WMMA GEMM --------------------------------------
// C[batch][64][N] = A[batch][64][K] (bf16 row-major) x BT[batch][N][K].
// Block = 256 threads = 8 waves; wave tile = 16(M) x 64(N); block = 64x128.
// grid.x = batches * N/128; grid.y = K-split chunks (KCHUNK_ = 1024 each).
// MODE 0: outF = acc                                 (S^T)
// MODE 1: outF = acc + aux0[b*64+m]*aux1[m*N+n]      (x_out)
// MODE 2: outBf = bf16(acc*aux0[m*N+n]+aux1[m*N+n])  (pred)
// MODE 3: outF[gridy-private partial] = acc          (t, K-split)
// BF32: B operand is f32 in memory, converted to bf16 in-register.
template <int MODE, bool BF32>
__global__ __launch_bounds__(256)
void wmma_gemm_kernel(const unsigned short* __restrict__ A, long aStride,
                      const void* __restrict__ Bptr, long bStride,
                      int N, int K,
                      float* __restrict__ outF,
                      unsigned short* __restrict__ outBf,
                      const float* __restrict__ aux0,
                      const float* __restrict__ aux1) {
  const int lane = threadIdx.x & 31;
  const int wave = threadIdx.x >> 5;
  const int nBlkPerBatch = N >> 7;
  const int batch = blockIdx.x / nBlkPerBatch;
  const int nblk  = blockIdx.x - batch * nBlkPerBatch;
  const int n0 = nblk * 128 + (wave >> 2) * 64;
  const int m0 = (wave & 3) * 16;
  const int kBegin = blockIdx.y * KCHUNK_;

  const int arow = m0 + (lane & 15);
  const int akb  = (lane & 16) ? 8 : 0;    // A k-offset within 32-chunk
  const int bkh  = (lane & 16) ? 8 : 0;    // B k-offset within 32-chunk
  const int bcol = lane & 15;

  const unsigned short* pA =
      A + (long)batch * aStride + (long)arow * K + kBegin + akb;

  v8f acc0, acc1, acc2, acc3;
#pragma unroll
  for (int r = 0; r < 8; ++r) { acc0[r] = 0.f; acc1[r] = 0.f;
                                acc2[r] = 0.f; acc3[r] = 0.f; }

  if (!BF32) {
    const unsigned short* Bb =
        (const unsigned short*)Bptr + (long)batch * bStride;
    const unsigned short* pB0 = Bb + (long)(n0 +  0 + bcol) * K + kBegin + bkh;
    const unsigned short* pB1 = Bb + (long)(n0 + 16 + bcol) * K + kBegin + bkh;
    const unsigned short* pB2 = Bb + (long)(n0 + 32 + bcol) * K + kBegin + bkh;
    const unsigned short* pB3 = Bb + (long)(n0 + 48 + bcol) * K + kBegin + bkh;
#pragma unroll 2
    for (int k = 0; k < KCHUNK_; k += 32) {
      Frag fa, b0, b1, b2, b3;
      loadFragBf(fa, pA  + k);
      loadFragBf(b0, pB0 + k);
      loadFragBf(b1, pB1 + k);
      loadFragBf(b2, pB2 + k);
      loadFragBf(b3, pB3 + k);
      __builtin_prefetch(pB0 + k + 256, 0, 1);   // global_prefetch_b8
      __builtin_prefetch(pB1 + k + 256, 0, 1);
      __builtin_prefetch(pB2 + k + 256, 0, 1);
      __builtin_prefetch(pB3 + k + 256, 0, 1);
      acc0 = __builtin_amdgcn_wmma_f32_16x16x32_bf16(false, fa.v, false, b0.v,
                                                     (short)0, acc0, false, false);
      acc1 = __builtin_amdgcn_wmma_f32_16x16x32_bf16(false, fa.v, false, b1.v,
                                                     (short)0, acc1, false, false);
      acc2 = __builtin_amdgcn_wmma_f32_16x16x32_bf16(false, fa.v, false, b2.v,
                                                     (short)0, acc2, false, false);
      acc3 = __builtin_amdgcn_wmma_f32_16x16x32_bf16(false, fa.v, false, b3.v,
                                                     (short)0, acc3, false, false);
    }
  } else {
    const float* Bf = (const float*)Bptr + (long)batch * bStride;
    const float* pB0 = Bf + (long)(n0 +  0 + bcol) * K + kBegin + bkh;
    const float* pB1 = Bf + (long)(n0 + 16 + bcol) * K + kBegin + bkh;
    const float* pB2 = Bf + (long)(n0 + 32 + bcol) * K + kBegin + bkh;
    const float* pB3 = Bf + (long)(n0 + 48 + bcol) * K + kBegin + bkh;
#pragma unroll 2
    for (int k = 0; k < KCHUNK_; k += 32) {
      Frag fa, b0, b1, b2, b3;
      loadFragBf (fa, pA  + k);
      loadFragF32(b0, pB0 + k);
      loadFragF32(b1, pB1 + k);
      loadFragF32(b2, pB2 + k);
      loadFragF32(b3, pB3 + k);
      acc0 = __builtin_amdgcn_wmma_f32_16x16x32_bf16(false, fa.v, false, b0.v,
                                                     (short)0, acc0, false, false);
      acc1 = __builtin_amdgcn_wmma_f32_16x16x32_bf16(false, fa.v, false, b1.v,
                                                     (short)0, acc1, false, false);
      acc2 = __builtin_amdgcn_wmma_f32_16x16x32_bf16(false, fa.v, false, b2.v,
                                                     (short)0, acc2, false, false);
      acc3 = __builtin_amdgcn_wmma_f32_16x16x32_bf16(false, fa.v, false, b3.v,
                                                     (short)0, acc3, false, false);
    }
  }

  const int crow = m0 + ((lane & 16) ? 8 : 0);
  const v8f* accs[4] = { &acc0, &acc1, &acc2, &acc3 };
#pragma unroll
  for (int t = 0; t < 4; ++t) {
    int n = n0 + t * 16 + bcol;
#pragma unroll
    for (int r = 0; r < 8; ++r) {
      int m = crow + r;
      float v = (*accs[t])[r];
      long o = ((long)batch * 64 + m) * N + n;
      if constexpr (MODE == 0) {
        outF[o] = v;
      } else if constexpr (MODE == 1) {
        outF[o] = v + aux0[batch * 64 + m] * aux1[m * N + n];
      } else if constexpr (MODE == 2) {
        outBf[o] = f2bf(v * aux0[m * N + n] + aux1[m * N + n]);
      } else {
        outF[(long)blockIdx.y * TPART_ + o] = v;
      }
    }
  }
}

// -------------------------- launcher ---------------------------------------

extern "C" void kernel_launch(void* const* d_in, const int* in_sizes, int n_in,
                              void* d_out, int out_size, void* d_ws, size_t ws_size,
                              hipStream_t stream) {
  (void)in_sizes; (void)n_in; (void)out_size; (void)ws_size;

  const float* x_inp   = (const float*)d_in[0];
  const float* W_A     = (const float*)d_in[1];
  const float* B_A     = (const float*)d_in[2];
  const float* W_F1    = (const float*)d_in[3];
  const float* W_F2    = (const float*)d_in[4];
  const float* B_F2    = (const float*)d_in[5];
  const float* W_G1    = (const float*)d_in[6];
  const float* W_G2    = (const float*)d_in[7];
  const float* B_G2    = (const float*)d_in[8];
  const float* W_S     = (const float*)d_in[9];
  const float* B_S     = (const float*)d_in[10];
  const float* betaU   = (const float*)d_in[11];
  const float* betaI   = (const float*)d_in[12];
  float* out = (float*)d_out;

  // workspace carve-up (256B aligned)
  char* p = (char*)d_ws; size_t off = 0;
  auto carve = [&](size_t bytes) {
    void* r = p + off; off = (off + bytes + 255) & ~(size_t)255; return r;
  };
  unsigned short* XsT   = (unsigned short*)carve((size_t)B_ * D_ * N_INP_ * 2);     // 64MB [b][d][i]
  unsigned short* phiT  = (unsigned short*)carve((size_t)B_ * N_OUT_ * N_INP_ * 2); // 4MB
  float* f_a            = (float*)carve((size_t)B_ * N_INP_ * 4);
  float* phisum         = (float*)carve((size_t)B_ * N_OUT_ * 4);
  float* t_part         = (float*)carve((size_t)4 * TPART_ * 4);                    // 8MB
  unsigned short* tw    = (unsigned short*)carve((size_t)B_ * N_OUT_ * D_ * 2);
  float* x_out          = (float*)carve((size_t)B_ * N_OUT_ * D_ * 4);
  unsigned short* lnbf  = (unsigned short*)carve((size_t)B_ * N_OUT_ * D_ * 2);
  unsigned short* pred  = (unsigned short*)carve((size_t)B_ * N_OUT_ * D_ * 2);
  float* SbufT          = (float*)carve((size_t)B_ * N_OUT_ * N_INP_ * 4);          // 8MB
  unsigned short* WF2T  = (unsigned short*)carve((size_t)D_ * D_ * 2);
  unsigned short* WG1T  = (unsigned short*)carve((size_t)D_ * D_ * 2);

  const long sPhiT = (long)N_OUT_ * N_INP_;
  const long sXsT  = (long)D_ * N_INP_;
  const long sJD   = (long)N_OUT_ * D_;

  // weight transposes (f32 -> bf16, N-major / K-contiguous)
  transpose_bf16_kernel<<<4096, 256, 0, stream>>>(W_F2, WF2T);   // [h][d]
  transpose_bf16_kernel<<<4096, 256, 0, stream>>>(W_G1, WG1T);   // [d][h]

  // scaled-x transpose + attention gate
  rowstats_kernel<<<B_ * N_INP_, 256, 0, stream>>>(x_inp, W_A, B_A, XsT, f_a);

  // ---- iteration 0 ----
  phi0_kernel<<<B_ * N_OUT_, 256, 0, stream>>>(f_a, betaU, betaI, phiT);
  phisum_kernel<<<B_ * N_OUT_, 256, 0, stream>>>(phiT, phisum);
  // t[b][j][d] = sum_i phiT[j][i] * Xs[i][d]   (K=4096, split 4)
  wmma_gemm_kernel<3, false><<<dim3(B_ * (D_ / 128), 4), 256, 0, stream>>>(
      phiT, sPhiT, XsT, sXsT, D_, N_INP_, t_part, nullptr, nullptr, nullptr);
  tw_kernel<<<(B_ * N_OUT_ * D_) / 256, 256, 0, stream>>>(t_part, W_F1, tw);
  // x_out = tw @ W_F2 + phisum*B_F2
  wmma_gemm_kernel<1, false><<<B_ * (D_ / 128), 256, 0, stream>>>(
      tw, sJD, WF2T, 0, D_, D_, x_out, nullptr, phisum, B_F2);

  // ---- routing update ----
  ln_kernel<<<B_ * N_OUT_, 256, 0, stream>>>(x_out, lnbf, nullptr);
  // pred[b][j][d] = (LN @ W_G1) * W_G2 + B_G2   (bf16)
  wmma_gemm_kernel<2, false><<<B_ * (D_ / 128), 256, 0, stream>>>(
      lnbf, sJD, WG1T, 0, D_, D_, nullptr, pred, W_G2, B_G2);
  // S^T[b][j][i] = sum_d pred[j][d] * x_inp[i][d]   (B operand f32, cvt in-reg)
  wmma_gemm_kernel<0, true><<<B_ * (N_INP_ / 128), 256, 0, stream>>>(
      pred, sJD, x_inp, (long)N_INP_ * D_, N_INP_, D_,
      SbufT, nullptr, nullptr, nullptr);
  softmax_phi_kernel<<<(B_ * N_INP_) / 8, 256, 0, stream>>>(
      SbufT, W_S, B_S, f_a, betaU, betaI, phiT);

  // ---- iteration 1 ----
  phisum_kernel<<<B_ * N_OUT_, 256, 0, stream>>>(phiT, phisum);
  wmma_gemm_kernel<3, false><<<dim3(B_ * (D_ / 128), 4), 256, 0, stream>>>(
      phiT, sPhiT, XsT, sXsT, D_, N_INP_, t_part, nullptr, nullptr, nullptr);
  tw_kernel<<<(B_ * N_OUT_ * D_) / 256, 256, 0, stream>>>(t_part, W_F1, tw);
  wmma_gemm_kernel<1, false><<<B_ * (D_ / 128), 256, 0, stream>>>(
      tw, sJD, WF2T, 0, D_, D_, x_out, nullptr, phisum, B_F2);

  // final LayerNorm -> output (f32)
  ln_kernel<<<B_ * N_OUT_, 256, 0, stream>>>(x_out, nullptr, out);
}
